// LaneRoI_32323923870243
// MI455X (gfx1250) — matrix operand order repaired and checked by reading, not compiled, not run on hardware
//
#include <hip/hip_runtime.h>

#define D    128
#define TILE 128
#define NTH  256
#define MATSZ (D * D)   // elements per 128x128 weight matrix

typedef __attribute__((ext_vector_type(8)))  float  v8f;
typedef __attribute__((ext_vector_type(8)))  __bf16 v8bf;
typedef __attribute__((ext_vector_type(16))) __bf16 v16bf;

__device__ __forceinline__ unsigned short f2bf(float f) {
  unsigned int u = __builtin_bit_cast(unsigned int, f);
  u += 0x7FFFu + ((u >> 16) & 1u);   // round-to-nearest-even
  return (unsigned short)(u >> 16);
}

// Async global -> LDS copy of 16 bytes (CDNA5 GLOBAL_LOAD_ASYNC_TO_LDS_B128,
// tracked by ASYNCcnt).  lds_byte = LDS byte offset (flat addr low 32 bits).
__device__ __forceinline__ void async_b128(unsigned lds_byte, const void* gptr) {
  asm volatile("global_load_async_to_lds_b128 %0, %1, off"
               :: "v"(lds_byte), "v"((unsigned long long)(size_t)gptr)
               : "memory");
}

__device__ __forceinline__ void wait_async_and_barrier() {
  asm volatile("s_wait_asynccnt 0" ::: "memory");
  __syncthreads();
}

// ---------------------------------------------------------------------------
// WMMA tile core.
// Alds: 128x128 bf16 row-major [M][K]   (gathered activations)
// Blds: 128x128 bf16 row-major [N][K]   (W pre-transposed -> contiguous-K frag)
// Each wave (8/block) computes rows [wave*16, wave*16+16) x all 128 cols.
// A frag (16-bit A 16x32): lanes0-15: M=lane,  K = 0..15
//                          lanes16-31: M=lane-16, K = 16..31
// B frag (16-bit B 32x16): lanes0-15: N=lane, K=0..15 ; lanes16-31: K=16..31
// ---------------------------------------------------------------------------
__device__ __forceinline__ void wmma_tile_core(const unsigned short* Alds,
                                               const unsigned short* Blds,
                                               int wave, int lane, v8f acc[8]) {
  const int mrow  = wave * 16 + (lane & 15);
  const int kbA   = (lane & 16) ? 16 : 0;
  const int kbB   = (lane & 16) ? 16 : 0;
  const int nlane = lane & 15;
#pragma unroll
  for (int kk = 0; kk < 4; ++kk) {
    const unsigned short* ap = &Alds[mrow * D + kk * 32 + kbA];
    v8bf alo = *(const v8bf*)ap;
    v8bf ahi = *(const v8bf*)(ap + 8);
    v16bf a = __builtin_shufflevector(alo, ahi,
                0, 1, 2, 3, 4, 5, 6, 7, 8, 9, 10, 11, 12, 13, 14, 15);
#pragma unroll
    for (int tn = 0; tn < 8; ++tn) {
      const int n = tn * 16 + nlane;
      v16bf b = *(const v16bf*)&Blds[n * D + kk * 32 + kbB];
      acc[tn] = __builtin_amdgcn_wmma_f32_16x16x32_bf16(
          false, a, false, b, (short)0, acc[tn], false, false);
    }
  }
}

// Stage preconverted weight matrix (32 KB contiguous) into LDS via async DMA.
__device__ __forceinline__ void stage_b_async(const unsigned short* WBmat,
                                              unsigned short* Blds, int tid) {
  const unsigned lds0 = (unsigned)(size_t)&Blds[0];
#pragma unroll
  for (int i = 0; i < 8; ++i) {
    const int off = (tid + i * NTH) * 16;   // byte offset, 16B chunks
    async_b128(lds0 + off, (const char*)WBmat + off);
  }
}

// ---------------------------------------------------------------------------
// One-time weight conversion: W [K][N] f32 -> WB [N][K] bf16 (one block/matrix)
// ---------------------------------------------------------------------------
__global__ __launch_bounds__(NTH)
void conv_w_kernel(const float* __restrict__ src, unsigned short* __restrict__ dst) {
  const float* S = src + (size_t)blockIdx.x * MATSZ;
  unsigned short* T = dst + (size_t)blockIdx.x * MATSZ;
  const int tid = threadIdx.x;
  const int k  = tid >> 1;
  const int c0 = (tid & 1) * 64;
  const float4* wr = (const float4*)&S[k * D + c0];
#pragma unroll
  for (int i = 0; i < 16; ++i) {
    float4 w = wr[i];
    const int n = c0 + i * 4;
    T[(n + 0) * D + k] = f2bf(w.x);
    T[(n + 1) * D + k] = f2bf(w.y);
    T[(n + 2) * D + k] = f2bf(w.z);
    T[(n + 3) * D + k] = f2bf(w.w);
  }
}

// ---------------------------------------------------------------------------
// Dense GEMM: Y[tile rows] = XB[tile rows] @ W  (plain f32 store)
// ---------------------------------------------------------------------------
__global__ __launch_bounds__(NTH)
void dense_gemm_kernel(const unsigned short* __restrict__ XB,
                       const unsigned short* __restrict__ WBmat,
                       float* __restrict__ Y, int Nn) {
  __shared__ __align__(32) unsigned short Alds[TILE * D];
  __shared__ __align__(32) unsigned short Blds[D * D];
  const int tid  = threadIdx.x;
  const int base = blockIdx.x * TILE;

  stage_b_async(WBmat, Blds, tid);
  {
    const int r  = tid >> 1;
    const int c0 = (tid & 1) * 64;          // 64 elements = 128 B
    int src = base + r; if (src >= Nn) src = Nn - 1;
    const unsigned short* g = &XB[(long)src * D + c0];
    const unsigned lds = (unsigned)(size_t)&Alds[r * D + c0];
#pragma unroll
    for (int i = 0; i < 8; ++i)
      async_b128(lds + i * 16, (const char*)g + i * 16);
  }
  wait_async_and_barrier();

  const int wave = tid >> 5, lane = tid & 31;
  v8f acc[8];
#pragma unroll
  for (int t = 0; t < 8; ++t)
#pragma unroll
    for (int e = 0; e < 8; ++e) acc[t][e] = 0.0f;

  wmma_tile_core(Alds, Blds, wave, lane, acc);

  const int nlo = lane & 15;
  const int mh  = (lane & 16) ? 8 : 0;
#pragma unroll
  for (int v = 0; v < 8; ++v) {
    const long row = (long)base + wave * 16 + v + mh;
    if (row < Nn) {
#pragma unroll
      for (int t = 0; t < 8; ++t)
        Y[row * D + t * 16 + nlo] = acc[t][v];
    }
  }
}

// ---------------------------------------------------------------------------
// Edge GEMM: TEMP[u[e]] += (XB[v[e]] @ W), 128-edge tiles, atomic f32 scatter.
// ---------------------------------------------------------------------------
__global__ __launch_bounds__(NTH)
void edge_gemm_kernel(const unsigned short* __restrict__ XB,
                      const unsigned short* __restrict__ WBmat,
                      const int* __restrict__ vIdx, const int* __restrict__ uIdx,
                      float* __restrict__ TEMP, int Ecnt) {
  __shared__ __align__(32) unsigned short Alds[TILE * D];
  __shared__ __align__(32) unsigned short Blds[D * D];
  const int tid  = threadIdx.x;
  const int base = blockIdx.x * TILE;

  stage_b_async(WBmat, Blds, tid);
  {
    const int r  = tid >> 1;
    const int c0 = (tid & 1) * 64;
    int e = base + r; if (e >= Ecnt) e = Ecnt - 1;
    const int src = vIdx[e];
    const unsigned short* g = &XB[(long)src * D + c0];
    const unsigned lds = (unsigned)(size_t)&Alds[r * D + c0];
#pragma unroll
    for (int i = 0; i < 8; ++i)
      async_b128(lds + i * 16, (const char*)g + i * 16);
  }
  wait_async_and_barrier();

  const int wave = tid >> 5, lane = tid & 31;
  v8f acc[8];
#pragma unroll
  for (int t = 0; t < 8; ++t)
#pragma unroll
    for (int e = 0; e < 8; ++e) acc[t][e] = 0.0f;

  wmma_tile_core(Alds, Blds, wave, lane, acc);

  const int nlo = lane & 15;
  const int mh  = (lane & 16) ? 8 : 0;
#pragma unroll
  for (int v = 0; v < 8; ++v) {
    const int edge = base + wave * 16 + v + mh;
    if (edge < Ecnt) {
      const long u = uIdx[edge];
      float* dst = &TEMP[u * D + nlo];
#pragma unroll
      for (int t = 0; t < 8; ++t)
        unsafeAtomicAdd(dst + t * 16, acc[t][v]);
    }
  }
}

// ---------------------------------------------------------------------------
// Row kernels: one wave per node row, 4 channels per lane (wave32 butterfly).
// ---------------------------------------------------------------------------
__device__ __forceinline__ void wave_mean_rstd(float4 xv, float& m, float& rs) {
  float sum = xv.x + xv.y + xv.z + xv.w;
  float sq  = xv.x * xv.x + xv.y * xv.y + xv.z * xv.z + xv.w * xv.w;
#pragma unroll
  for (int off = 16; off > 0; off >>= 1) {
    sum += __shfl_xor(sum, off, 32);
    sq  += __shfl_xor(sq,  off, 32);
  }
  m = sum * (1.0f / 128.0f);
  float var = sq * (1.0f / 128.0f) - m * m;
  rs = rsqrtf(var + 1e-5f);
}

__device__ __forceinline__ void store_bf16x4(unsigned short* p, float4 y) {
  ushort4 h;
  h.x = f2bf(y.x); h.y = f2bf(y.y); h.z = f2bf(y.z); h.w = f2bf(y.w);
  *(ushort4*)p = h;
}

// x = relu(GN(temp)) -> bf16 XB only (f32 copy not needed downstream)
__global__ __launch_bounds__(256)
void gn_relu_kernel(const float* __restrict__ T, const float* __restrict__ s,
                    const float* __restrict__ b, unsigned short* __restrict__ XB,
                    int Nn) {
  const int wave = threadIdx.x >> 5, lane = threadIdx.x & 31;
  const long row = (long)blockIdx.x * 8 + wave;
  if (row >= Nn) return;
  const int c0 = lane * 4;
  float4 xv = *(const float4*)&T[row * D + c0];
  float m, rs;
  wave_mean_rstd(xv, m, rs);
  float4 sv = *(const float4*)&s[c0];
  float4 bv = *(const float4*)&b[c0];
  float4 y;
  y.x = fmaxf((xv.x - m) * rs * sv.x + bv.x, 0.0f);
  y.y = fmaxf((xv.y - m) * rs * sv.y + bv.y, 0.0f);
  y.z = fmaxf((xv.z - m) * rs * sv.z + bv.z, 0.0f);
  y.w = fmaxf((xv.w - m) * rs * sv.w + bv.w, 0.0f);
  store_bf16x4(&XB[row * D + c0], y);
}

// x = relu(GN(z) + identity); identity = x.  Writes f32 X (=d_out), IDENT, XB.
__global__ __launch_bounds__(256)
void gn_resid_kernel(const float* __restrict__ Z, const float* __restrict__ s,
                     const float* __restrict__ b, float* __restrict__ IDENT,
                     float* __restrict__ X, unsigned short* __restrict__ XB,
                     int Nn) {
  const int wave = threadIdx.x >> 5, lane = threadIdx.x & 31;
  const long row = (long)blockIdx.x * 8 + wave;
  if (row >= Nn) return;
  const int c0 = lane * 4;
  float4 xv = *(const float4*)&Z[row * D + c0];
  float m, rs;
  wave_mean_rstd(xv, m, rs);
  float4 sv = *(const float4*)&s[c0];
  float4 bv = *(const float4*)&b[c0];
  float4 iv = *(const float4*)&IDENT[row * D + c0];
  float4 y;
  y.x = fmaxf((xv.x - m) * rs * sv.x + bv.x + iv.x, 0.0f);
  y.y = fmaxf((xv.y - m) * rs * sv.y + bv.y + iv.y, 0.0f);
  y.z = fmaxf((xv.z - m) * rs * sv.z + bv.z + iv.z, 0.0f);
  y.w = fmaxf((xv.w - m) * rs * sv.w + bv.w + iv.w, 0.0f);
  *(float4*)&X[row * D + c0]     = y;
  *(float4*)&IDENT[row * D + c0] = y;
  store_bf16x4(&XB[row * D + c0], y);
}

// input: relu(GN(feat @ input_w)) -> IDENT (f32) + XB (bf16)
__global__ __launch_bounds__(256)
void input_kernel(const float* __restrict__ feat, const float* __restrict__ Wi,
                  const float* __restrict__ s, const float* __restrict__ b,
                  float* __restrict__ IDENT, unsigned short* __restrict__ XB,
                  int Nn) {
  const int wave = threadIdx.x >> 5, lane = threadIdx.x & 31;
  const long row = (long)blockIdx.x * 8 + wave;
  if (row >= Nn) return;
  float f[16];
  const float4* fr = (const float4*)&feat[row * 16];
#pragma unroll
  for (int i = 0; i < 4; ++i) {
    float4 t = fr[i];
    f[i * 4 + 0] = t.x; f[i * 4 + 1] = t.y; f[i * 4 + 2] = t.z; f[i * 4 + 3] = t.w;
  }
  const int c0 = lane * 4;
  float4 acc = make_float4(0.f, 0.f, 0.f, 0.f);
#pragma unroll
  for (int k = 0; k < 16; ++k) {
    float4 w = *(const float4*)&Wi[k * D + c0];
    acc.x = fmaf(f[k], w.x, acc.x);
    acc.y = fmaf(f[k], w.y, acc.y);
    acc.z = fmaf(f[k], w.z, acc.z);
    acc.w = fmaf(f[k], w.w, acc.w);
  }
  float m, rs;
  wave_mean_rstd(acc, m, rs);
  float4 sv = *(const float4*)&s[c0];
  float4 bv = *(const float4*)&b[c0];
  float4 y;
  y.x = fmaxf((acc.x - m) * rs * sv.x + bv.x, 0.0f);
  y.y = fmaxf((acc.y - m) * rs * sv.y + bv.y, 0.0f);
  y.z = fmaxf((acc.z - m) * rs * sv.z + bv.z, 0.0f);
  y.w = fmaxf((acc.w - m) * rs * sv.w + bv.w, 0.0f);
  *(float4*)&IDENT[row * D + c0] = y;
  store_bf16x4(&XB[row * D + c0], y);
}

// ---------------------------------------------------------------------------
extern "C" void kernel_launch(void* const* d_in, const int* in_sizes, int n_in,
                              void* d_out, int out_size, void* d_ws, size_t ws_size,
                              hipStream_t stream) {
  const float* feat     = (const float*)d_in[0];
  const float* input_w  = (const float*)d_in[1];
  const float* in_gn_s  = (const float*)d_in[2];
  const float* in_gn_b  = (const float*)d_in[3];
  const float* ctr_w    = (const float*)d_in[4];
  const float* pre_w    = (const float*)d_in[5];
  const float* suc_w    = (const float*)d_in[6];
  const float* left_w   = (const float*)d_in[7];
  const float* right_w  = (const float*)d_in[8];
  const float* norm_s   = (const float*)d_in[9];
  const float* norm_b   = (const float*)d_in[10];
  const float* ctr2_w   = (const float*)d_in[11];
  const float* c2_gn_s  = (const float*)d_in[12];
  const float* c2_gn_b  = (const float*)d_in[13];
  const int*   pre_u    = (const int*)d_in[14];
  const int*   pre_v    = (const int*)d_in[15];
  const int*   suc_u    = (const int*)d_in[16];
  const int*   suc_v    = (const int*)d_in[17];
  const int*   left_u   = (const int*)d_in[18];
  const int*   left_v   = (const int*)d_in[19];
  const int*   right_u  = (const int*)d_in[20];
  const int*   right_v  = (const int*)d_in[21];

  const int N  = in_sizes[0] / 16;
  const int E  = in_sizes[14] / 6;   // pre_u is (S=6, E)
  const int E2 = in_sizes[18];
  const int S  = 6;
  const size_t NN = (size_t)N * D;

  float*          X     = (float*)d_out;                       // final output
  float*          TEMP  = (float*)d_ws;                        // agg / Z buffer
  float*          IDENT = TEMP + NN;                           // residual
  unsigned short* XB    = (unsigned short*)(IDENT + NN);       // bf16 activations
  unsigned short* WB    = XB + NN;                             // bf16 weights

  // WB layout (matrices of MATSZ bf16): [ctr:4][pre:24][suc:24][left:4][right:4][ctr2:4]
  const size_t CTR_OFF = 0, PRE_OFF = 4, SUC_OFF = 28, LEFT_OFF = 52,
               RIGHT_OFF = 56, CTR2_OFF = 60;

  const int gnBlocks = (N + 7) / 8;
  const int tilesN   = (N + TILE - 1) / TILE;
  const int tilesE   = (E + TILE - 1) / TILE;
  const int tilesE2  = (E2 + TILE - 1) / TILE;

  // one-time weight conversion (f32 [K][N] -> bf16 [N][K])
  conv_w_kernel<<<4,  NTH, 0, stream>>>(ctr_w,   WB + CTR_OFF  * MATSZ);
  conv_w_kernel<<<24, NTH, 0, stream>>>(pre_w,   WB + PRE_OFF  * MATSZ);
  conv_w_kernel<<<24, NTH, 0, stream>>>(suc_w,   WB + SUC_OFF  * MATSZ);
  conv_w_kernel<<<4,  NTH, 0, stream>>>(left_w,  WB + LEFT_OFF * MATSZ);
  conv_w_kernel<<<4,  NTH, 0, stream>>>(right_w, WB + RIGHT_OFF* MATSZ);
  conv_w_kernel<<<4,  NTH, 0, stream>>>(ctr2_w,  WB + CTR2_OFF * MATSZ);

  input_kernel<<<gnBlocks, 256, 0, stream>>>(feat, input_w, in_gn_s, in_gn_b,
                                             IDENT, XB, N);

  for (int i = 0; i < 4; ++i) {
    // temp = x @ ctr_w[i]
    dense_gemm_kernel<<<tilesN, NTH, 0, stream>>>(
        XB, WB + (CTR_OFF + i) * MATSZ, TEMP, N);
    // scatter-add relations
    for (int h = 0; h < S; ++h) {
      edge_gemm_kernel<<<tilesE, NTH, 0, stream>>>(
          XB, WB + (PRE_OFF + (size_t)i * S + h) * MATSZ,
          pre_v + (size_t)h * E, pre_u + (size_t)h * E, TEMP, E);
      edge_gemm_kernel<<<tilesE, NTH, 0, stream>>>(
          XB, WB + (SUC_OFF + (size_t)i * S + h) * MATSZ,
          suc_v + (size_t)h * E, suc_u + (size_t)h * E, TEMP, E);
    }
    edge_gemm_kernel<<<tilesE2, NTH, 0, stream>>>(
        XB, WB + (LEFT_OFF + i) * MATSZ, left_v, left_u, TEMP, E2);
    edge_gemm_kernel<<<tilesE2, NTH, 0, stream>>>(
        XB, WB + (RIGHT_OFF + i) * MATSZ, right_v, right_u, TEMP, E2);

    // x = relu(GN(temp))  -> bf16
    gn_relu_kernel<<<gnBlocks, 256, 0, stream>>>(
        TEMP, norm_s + (size_t)i * D, norm_b + (size_t)i * D, XB, N);
    // z = x @ ctr2_w[i]   (reuse TEMP as Z)
    dense_gemm_kernel<<<tilesN, NTH, 0, stream>>>(
        XB, WB + (CTR2_OFF + i) * MATSZ, TEMP, N);
    // x = relu(GN(z) + identity); identity = x
    gn_resid_kernel<<<gnBlocks, 256, 0, stream>>>(
        TEMP, c2_gn_s + (size_t)i * D, c2_gn_b + (size_t)i * D,
        IDENT, X, XB, N);
  }
}
